// SparseGraphGen_14499809591643
// MI455X (gfx1250) — matrix operand order repaired and checked by reading, not compiled, not run on hardware
//
#include <hip/hip_runtime.h>
#include <hip/hip_bf16.h>

// SparseGraphGen on MI455X (gfx1250, wave32, WMMA, async global->LDS).
// B=8, N=2048, H=F_IN=64, K=50, 2 message-passing iterations.

#define B_ 8
#define N_ 2048
#define H_ 64
#define K_ 50

typedef __attribute__((ext_vector_type(16))) __bf16 v16bf;
typedef __attribute__((ext_vector_type(8)))  float  v8f;
typedef __attribute__((ext_vector_type(4)))  int    v4i;

union BF16x16 { unsigned short u[16]; v16bf v; };

// ---- gfx1250 async global->LDS (guarded: falls back to sync copy if the
// toolchain lacks the builtins; __has_builtin keeps both passes compiling).
#if defined(__AMDGCN__) && \
    __has_builtin(__builtin_amdgcn_global_load_async_to_lds_b128) && \
    __has_builtin(__builtin_amdgcn_s_wait_asynccnt)
#define SG_HAVE_ASYNC_LDS 1
#else
#define SG_HAVE_ASYNC_LDS 0
#endif

#define SG_AS1 __attribute__((address_space(1)))
#define SG_AS3 __attribute__((address_space(3)))

__device__ __forceinline__ unsigned short f32_to_bf16(float f) {
  unsigned int u = __float_as_uint(f);
  u += 0x7FFFu + ((u >> 16) & 1u);           // round-to-nearest-even
  return (unsigned short)(u >> 16);
}

// Copy one 256-row x 64-col bf16 super-tile (32 KiB) global -> LDS.
// 2048 16-byte chunks, 8 per thread (256-thread block).
__device__ __forceinline__ void sg_stage(const unsigned short* __restrict__ gsrc,
                                         unsigned short* __restrict__ ldst,
                                         int tid) {
#if SG_HAVE_ASYNC_LDS
#pragma unroll
  for (int j = 0; j < 8; ++j) {
    const char* g = (const char*)gsrc + (size_t)(tid + j * 256) * 16;
    char*       l = (char*)ldst + (size_t)(tid + j * 256) * 16;
    // generic->AS1: identical address; generic->AS3: LDS offset is the low
    // 32 bits of the generic pointer (LDS aperture base has zero low bits).
    __builtin_amdgcn_global_load_async_to_lds_b128(
        (SG_AS1 v4i*)(unsigned long long)g,
        (SG_AS3 v4i*)(unsigned int)(unsigned long long)l,
        0, 0);
  }
#else
  const uint4* s = (const uint4*)gsrc;
  uint4*       d = (uint4*)ldst;
#pragma unroll
  for (int j = 0; j < 8; ++j) d[tid + j * 256] = s[tid + j * 256];
#endif
}

__device__ __forceinline__ void sg_stage_wait() {
#if SG_HAVE_ASYNC_LDS
  __builtin_amdgcn_s_wait_asynccnt(0);       // own wave's async DMAs done
#endif
}

// ---------------------------------------------------------------------------
// Y[row,:] = relu( concat(X1[row,0:d1], X2[row,0:d2]) @ W + bias ), H_=64 out.
// One row per 64-thread block; W is row-major [din, 64] (jax C-order), so
// W[d*64 + tid] is coalesced across the 64 lanes.
// ---------------------------------------------------------------------------
__global__ void sg_mlp64_relu(const float* __restrict__ X1, int d1,
                              const float* __restrict__ X2, int d2,
                              const float* __restrict__ W,
                              const float* __restrict__ bias,
                              float* __restrict__ Y,
                              unsigned short* __restrict__ Ybf) {
  __shared__ float xrow[128];
  const int row = blockIdx.x;
  const int tid = threadIdx.x;                 // 0..63
  const int din = d1 + d2;
  if (tid < d1) xrow[tid] = X1[(long)row * d1 + tid];
  if (d2 > 0 && tid < d2) xrow[d1 + tid] = X2[(long)row * d2 + tid];
  __syncthreads();
  float acc = bias[tid];
#pragma unroll 16
  for (int d = 0; d < din; ++d) acc = fmaf(xrow[d], W[d * H_ + tid], acc);
  acc = fmaxf(acc, 0.0f);
  Y[(long)row * H_ + tid] = acc;
  if (Ybf) Ybf[(long)row * H_ + tid] = f32_to_bf16(acc);
}

// ---------------------------------------------------------------------------
// Fused S = h.h^T tile + per-row top-K.  One 256-thread WG (8 waves) handles
// (batch b, 16-row tile).  Full 16 x 2048 f32 score strip lives in LDS
// (128 KiB -- CDNA5 WGP has 320 KB).  Column h data double-buffered in LDS,
// staged 256 rows at a time with async global->LDS DMA overlapping WMMA.
// LDS: scores f32[16*N_] | hA u16[16*64] | hB0 u16[256*64] | hB1 u16[256*64]
// ---------------------------------------------------------------------------
#define SIMTOPK_LDS_BYTES (16 * N_ * 4 + 16 * H_ * 2 + 2 * 256 * H_ * 2)

__global__ void sg_sim_topk(const unsigned short* __restrict__ hbf,
                            float* __restrict__ out_vals,
                            int* __restrict__ out_idx) {
  extern __shared__ char smem[];
  float*          scores = (float*)smem;
  unsigned short* hA     = (unsigned short*)(smem + 16 * N_ * 4);
  unsigned short* hB0    = (unsigned short*)(smem + 16 * N_ * 4 + 16 * H_ * 2);
  unsigned short* hB1    = hB0 + 256 * H_;

  const int tid     = threadIdx.x;             // 0..255
  const int lane    = tid & 31;
  const int wave    = tid >> 5;                // 0..7
  const int rt      = blockIdx.x & (N_ / 16 - 1);
  const int b       = blockIdx.x / (N_ / 16);
  const int rowbase = rt * 16;

  const unsigned short* hbatch = hbf + (size_t)b * N_ * H_;

  // ---- stage A tile (16 rows x 64 bf16, contiguous) into LDS
  {
    const unsigned int* src = (const unsigned int*)(hbatch + (size_t)rowbase * H_);
    unsigned int* dst = (unsigned int*)hA;
    for (int j = tid; j < 16 * H_ / 2; j += 256) dst[j] = src[j];
  }

  // ---- kick off async stage of super-tile 0
  sg_stage(hbatch, hB0, tid);
  sg_stage_wait();
  __syncthreads();

  // ---- per-lane A fragments (contiguous k order, same as B):
  // lane<16: row=lane, K = kb+0..15; lane>=16: row=lane-16, K = kb+16..31.
  BF16x16 a0, a1;
  {
    const int arow = lane & 15;
    const int aoff = (lane < 16) ? 0 : 16;
#pragma unroll
    for (int i = 0; i < 16; ++i) {
      a0.u[i]     = hA[arow * H_ + 0  + aoff + i];
      a1.u[i]     = hA[arow * H_ + 32 + aoff + i];
    }
  }

  // ---- sweep 2048 columns in 8 super-tiles of 256, double-buffered
  for (int st = 0; st < N_ / 256; ++st) {
    unsigned short* cur = (st & 1) ? hB1 : hB0;
    unsigned short* nxt = (st & 1) ? hB0 : hB1;
    if (st + 1 < N_ / 256)                     // DMA next stage during WMMA
      sg_stage(hbatch + (size_t)(st + 1) * 256 * H_, nxt, tid);

    // each wave: 2 column tiles of 16  (8 waves x 2 x 16 = 256 columns)
#pragma unroll
    for (int t = 0; t < 2; ++t) {
      const int ct      = wave * 2 + t;        // 0..15 within super-tile
      const int colbase = ct * 16;
      // B fragments (contiguous k order, same as A): lane<16 -> K=kb+0..15, col=lane;
      // lane>=16 -> K=kb+16..31, col=lane-16.  16 contiguous halves per lane.
      BF16x16 b0f, b1f;
      const int bcol = lane & 15;
      const int koff = (lane < 16) ? 0 : 16;
#pragma unroll
      for (int i = 0; i < 16; ++i) {
        b0f.u[i] = cur[(colbase + bcol) * H_ + 0  + koff + i];
        b1f.u[i] = cur[(colbase + bcol) * H_ + 32 + koff + i];
      }
      v8f c = {};
      c = __builtin_amdgcn_wmma_f32_16x16x32_bf16(false, a0.v, false, b0f.v,
                                                  (short)0, c, false, false);
      c = __builtin_amdgcn_wmma_f32_16x16x32_bf16(false, a1.v, false, b1f.v,
                                                  (short)0, c, false, false);
      // D layout: lane<16 -> N=lane, M=vgpr r; lane>=16 -> N=lane-16, M=8+r.
      const int gcol  = st * 256 + colbase + bcol;
      const int mbase = (lane < 16) ? 0 : 8;
#pragma unroll
      for (int r = 0; r < 8; ++r) scores[(mbase + r) * N_ + gcol] = c[r];
    }

    sg_stage_wait();                           // next buffer resident
    __syncthreads();                           // all waves done with cur
  }

  // ---- per-row top-K via 50 wave-wide argmax extractions (2 rows / wave)
  for (int rr = 0; rr < 2; ++rr) {
    const int r = wave * 2 + rr;
    float* srow = scores + r * N_;
    const long obase = ((long)(b * N_ + rowbase + r)) * K_;
    for (int k = 0; k < K_; ++k) {
      float best = -3.4e38f;
      int   bidx = 0;
      for (int i = 0; i < N_ / 32; ++i) {
        const int col = lane + i * 32;
        const float v = srow[col];
        if (v > best) { best = v; bidx = col; }
      }
#pragma unroll
      for (int d = 16; d > 0; d >>= 1) {
        const float ov = __shfl_xor(best, d, 32);
        const int   oi = __shfl_xor(bidx, d, 32);
        if (ov > best || (ov == best && oi < bidx)) { best = ov; bidx = oi; }
      }
      if (lane == 0) {
        out_vals[obase + k] = best;
        out_idx[obase + k]  = bidx;
        srow[bidx] = -3.4e38f;                 // mask winner
      }
      __builtin_amdgcn_wave_barrier();
      asm volatile("s_wait_dscnt 0" ::: "memory");
      __builtin_amdgcn_wave_barrier();
    }
  }
}

// ---------------------------------------------------------------------------
// agg[row,:] = sum_k vals[row,k] * msg[b, idx[row,k], :]
// ---------------------------------------------------------------------------
__global__ void sg_aggregate(const float* __restrict__ vals,
                             const int* __restrict__ idx,
                             const float* __restrict__ msg,
                             float* __restrict__ agg) {
  const int row = blockIdx.x;                  // 0..B*N-1
  const int tid = threadIdx.x;                 // 0..63
  const int b   = row / N_;
  const long vb = (long)row * K_;
  float acc = 0.f;
  for (int k = 0; k < K_; ++k) {
    const float v = vals[vb + k];
    const int   j = idx[vb + k];
    acc = fmaf(v, msg[((long)(b * N_ + j)) * H_ + tid], acc);
  }
  agg[(long)row * H_ + tid] = acc;
}

__global__ void sg_zero4(float4* __restrict__ p, long n4) {
  const long i = (long)blockIdx.x * blockDim.x + threadIdx.x;
  if (i < n4) p[i] = make_float4(0.f, 0.f, 0.f, 0.f);
}

__global__ void sg_scatter(const float* __restrict__ vals,
                           const int* __restrict__ idx,
                           float* __restrict__ out) {
  const long t = (long)blockIdx.x * blockDim.x + threadIdx.x;
  if (t >= (long)B_ * N_ * K_) return;
  const long row = t / K_;                     // global row (b*N + n)
  out[row * N_ + idx[t]] = vals[t];
}

// ---------------------------------------------------------------------------
extern "C" void kernel_launch(void* const* d_in, const int* in_sizes, int n_in,
                              void* d_out, int out_size, void* d_ws, size_t ws_size,
                              hipStream_t stream) {
  (void)in_sizes; (void)n_in; (void)out_size; (void)ws_size;
  const float* x   = (const float*)d_in[0];
  const float* ew0 = (const float*)d_in[1];
  const float* eb0 = (const float*)d_in[2];
  const float* ew1 = (const float*)d_in[3];
  const float* eb1 = (const float*)d_in[4];
  const float* wm[2] = {(const float*)d_in[5], (const float*)d_in[9]};
  const float* bm[2] = {(const float*)d_in[6], (const float*)d_in[10]};
  const float* wu[2] = {(const float*)d_in[7], (const float*)d_in[11]};
  const float* bu[2] = {(const float*)d_in[8], (const float*)d_in[12]};

  // workspace layout (bytes): ~25.2 MiB total
  char* ws = (char*)d_ws;
  float*          hPing = (float*)(ws);                          // 4 MiB
  float*          hPong = (float*)(ws + (4u  << 20));            // 4 MiB
  unsigned short* hbf   = (unsigned short*)(ws + (8u  << 20));   // 2 MiB
  float*          tmp   = (float*)(ws + (10u << 20));            // 4 MiB
  float*          agg   = (float*)(ws + (14u << 20));            // 4 MiB
  float*          vals  = (float*)(ws + (18u << 20));            // 3.2 MiB
  int*            idx   = (int*)  (ws + (22u << 20));            // 3.2 MiB

  const int ROWS = B_ * N_;

  // node embedding: two fused GEMM+ReLU layers; second also emits bf16 h
  sg_mlp64_relu<<<ROWS, 64, 0, stream>>>(x,   64, nullptr, 0, ew0, eb0, tmp,   nullptr);
  sg_mlp64_relu<<<ROWS, 64, 0, stream>>>(tmp, 64, nullptr, 0, ew1, eb1, hPing, hbf);

  float* hcur = hPing; float* hnext = hPong;
  for (int it = 0; it < 2; ++it) {
    sg_sim_topk<<<B_ * (N_ / 16), 256, SIMTOPK_LDS_BYTES, stream>>>(hbf, vals, idx);
    sg_mlp64_relu<<<ROWS, 64, 0, stream>>>(hcur, 64, nullptr, 0, wm[it], bm[it], tmp, nullptr);
    sg_aggregate<<<ROWS, 64, 0, stream>>>(vals, idx, tmp, agg);
    sg_mlp64_relu<<<ROWS, 64, 0, stream>>>(hcur, 64, agg, 64, wu[it], bu[it], hnext, hbf);
    float* t = hcur; hcur = hnext; hnext = t;
  }

  // final dense top-k adjacency
  sg_sim_topk<<<B_ * (N_ / 16), 256, SIMTOPK_LDS_BYTES, stream>>>(hbf, vals, idx);

  const long n4 = (long)B_ * N_ * N_ / 4;
  sg_zero4<<<(unsigned)((n4 + 255) / 256), 256, 0, stream>>>((float4*)d_out, n4);
  sg_scatter<<<(B_ * N_ * K_ + 255) / 256, 256, 0, stream>>>(vals, idx, (float*)d_out);
}
